// DGCNN_reg_5686536700341
// MI455X (gfx1250) — matrix operand order repaired and checked by reading, not compiled, not run on hardware
//
#include <hip/hip_runtime.h>
#include <hip/hip_bf16.h>
#include <math.h>

// DGCNN forward for MI455X (gfx1250, wave32, WMMA), fragment-swizzled operands.
// B=16, N=2048, K=20.
//  - conv GEMMs: v_wmma_f32_16x16x32_bf16 (f32 accum), A/B pre-swizzled so each
//    fragment is 32 contiguous bytes per lane (global_load_b128 / ds_load_b128).
//  - knn inner products: v_wmma_f32_16x16x4_f32 on swizzled float2 fragments.
//  - 320KB/WGP LDS holds a full 16x2048 distance row-block for top-20 selection.

#define BATCH 16
#define NPTS  2048
#define KNN   20

typedef __attribute__((ext_vector_type(16))) __bf16 v16bf;
typedef __attribute__((ext_vector_type(8)))  float  v8f;
typedef __attribute__((ext_vector_type(2)))  float  v2f;

__device__ __forceinline__ float lrelu(float v) { return v > 0.f ? v : 0.2f * v; }

// monotone float<->u32 encoding so atomicMax(u32) == float max (handles negatives)
__device__ __forceinline__ unsigned fenc(float f) {
  unsigned u = __float_as_uint(f);
  return (u & 0x80000000u) ? ~u : (u | 0x80000000u);
}
__device__ __forceinline__ float fdec(unsigned e) {
  unsigned u = (e & 0x80000000u) ? (e ^ 0x80000000u) : ~e;
  return __uint_as_float(u);
}

// Fragment-order storage: tile t, lane L -> 16 contiguous bf16 at (t*32+L)*16.
// A-tile (16x32) element e of lane L=(m + 16*h): k = h*16+e
// B-tile (32x16) element e of lane L=(n + 16*h): k = h*16+e
__device__ __forceinline__ v16bf load_frag16(const __bf16* p, int tileIdx) {
  return *(const v16bf*)(p + ((size_t)tileIdx * 32 + (threadIdx.x & 31)) * 16);
}

// scatter position helpers (inverse maps, used when *writing* swizzled panels)
__device__ __forceinline__ size_t a_pos(int tileIdx, int m, int kin) {
  int h = (kin >> 4) & 1;
  int e = kin & 15;
  return ((size_t)tileIdx * 32 + (m + 16 * h)) * 16 + e;
}
__device__ __forceinline__ size_t b_pos(int tileIdx, int n, int kin) {
  return ((size_t)tileIdx * 32 + ((kin >> 4) * 16 + n)) * 16 + (kin & 15);
}

// ---------------- weight fp32 -> bf16, swizzled to A-fragment order ----------------
__global__ void k_convert_w_swz(const float* __restrict__ src, __bf16* __restrict__ dst,
                                int O, int C, int Cpad) {
  int i = blockIdx.x * blockDim.x + threadIdx.x;
  if (i >= O * Cpad) return;
  int o = i / Cpad, c = i % Cpad;
  int numKc = Cpad >> 5;
  size_t pos = a_pos((o >> 4) * numKc + (c >> 5), o & 15, c & 31);
  dst[pos] = (c < C) ? (__bf16)src[o * C + c] : (__bf16)0.f;
}

__global__ void k_init_g(unsigned* g) {
  int i = blockIdx.x * blockDim.x + threadIdx.x;
  if (i < BATCH * 1024) g[i] = 0u;  // == fenc(very negative)
}

// ---------------- swizzle x into f32-WMMA fragment order + precompute ||x||^2 ----
// f32 16x16x4: A and B fragments share the same (lane,elem)->(k, idx) map:
//   lane L=(i + 16*h), elem e in v2f: k = 2h+e. Tile ct covers 16 points.
template <int C>
__global__ void k_swz_x(const float* __restrict__ X, float* __restrict__ Xswz) {
  constexpr int CP = (C + 3) & ~3;
  constexpr int NKC = CP / 4;
  constexpr int PER = (NPTS / 16) * NKC * 64;  // floats per batch
  int b = blockIdx.y;
  int i = blockIdx.x * blockDim.x + threadIdx.x;
  if (i >= PER) return;
  int e = i & 1, lane = (i >> 1) & 31;
  int rest = i >> 6, kc = rest % NKC, ct = rest / NKC;
  int k = kc * 4 + 2 * (lane >> 4) + e;
  int n = ct * 16 + (lane & 15);
  float v = (k < C) ? X[((size_t)b * C + k) * NPTS + n] : 0.f;
  Xswz[(size_t)b * PER + i] = v;
}

__global__ void k_xx(const float* __restrict__ X, int C, float* __restrict__ xx) {
  int i = blockIdx.x * blockDim.x + threadIdx.x;
  if (i >= BATCH * NPTS) return;
  int b = i / NPTS, n = i % NPTS;
  float s = 0.f;
  for (int c = 0; c < C; ++c) { float v = X[((size_t)b * C + c) * NPTS + n]; s += v * v; }
  xx[i] = s;
}

// ---------------- knn: top-20 of pd = 2*x^T x - xx_n - xx_m ----------------
// block = 256 thr (8 waves) owns 16 query rows of batch b.
// Phase 1: 16x16 distance tiles via v_wmma_f32_16x16x4_f32; row fragments are
//          hoisted into registers; column fragments are single global_load_b64.
// Phase 2: iterative argmax top-20, 2 rows per wave, in the 128KB LDS row-block.
template <int C>
__global__ void k_knn(const float* __restrict__ Xswz, const float* __restrict__ xx,
                      int* __restrict__ idx) {
  constexpr int CP = (C + 3) & ~3;
  constexpr int NKC = CP / 4;
  constexpr int PER = (NPTS / 16) * NKC * 64;
  extern __shared__ __align__(32) char smem[];
  float* dist = (float*)smem;       // [16][NPTS]
  float* xxRow = dist + 16 * NPTS;  // [16]

  const int b = blockIdx.y, n0 = blockIdx.x * 16;
  const int tid = threadIdx.x, lane = tid & 31, wave = tid >> 5;
  const float* Xs = Xswz + (size_t)b * PER;
  const float* xxb = xx + (size_t)b * NPTS;
  if (tid < 16) xxRow[tid] = xxb[n0 + tid];
  __syncthreads();

  const int h = lane >> 4, nl = lane & 15;
  const int rowCt = n0 >> 4;
  v2f afrag[NKC];
#pragma unroll
  for (int kc = 0; kc < NKC; ++kc)
    afrag[kc] = *(const v2f*)(Xs + ((size_t)(rowCt * NKC + kc) * 32 + lane) * 2);

  for (int ct = wave; ct < NPTS / 16; ct += (blockDim.x >> 5)) {
    v8f acc = {};
#pragma unroll
    for (int kc = 0; kc < NKC; ++kc) {
      v2f bb = *(const v2f*)(Xs + ((size_t)(ct * NKC + kc) * 32 + lane) * 2);
      acc = __builtin_amdgcn_wmma_f32_16x16x4_f32(false, afrag[kc], false, bb, (short)0,
                                                  acc, false, false);
    }
    int col = ct * 16 + nl;
    float xc = xxb[col];
#pragma unroll
    for (int r = 0; r < 8; ++r) {
      int m = r + 8 * h;
      dist[m * NPTS + col] = 2.f * acc[r] - xxRow[m] - xc;
    }
  }
  __syncthreads();

  // top-20 per row; wave w owns rows 2w, 2w+1 (uniform loops => uniform barriers)
  for (int rr = 0; rr < 2; ++rr) {
    int m = wave * 2 + rr;
    float* drow = dist + m * NPTS;
    for (int kk = 0; kk < KNN; ++kk) {
      float best = -1e30f; int bi = 0;
      for (int j = lane; j < NPTS; j += 32) {
        float v = drow[j];
        if (v > best) { best = v; bi = j; }
      }
#pragma unroll
      for (int off = 16; off; off >>= 1) {
        float ov = __shfl_xor(best, off, 32);
        int   oi = __shfl_xor(bi, off, 32);
        if (ov > best || (ov == best && oi < bi)) { best = ov; bi = oi; }
      }
      if (lane == 0) {
        idx[((size_t)b * NPTS + n0 + m) * KNN + kk] = bi;
        drow[bi] = -3e38f;
      }
      __syncthreads();
    }
  }
}

// ---------------- fused edge-conv stage ----------------
// gather (nbr-ctr, ctr) into a B-fragment-ordered bf16 panel -> conv wA ->
// [conv wB] -> max over K=20. Block owns (b, 16 points) => 320 "nk" columns.
template <int CIN, int KA, bool TWO>
__global__ void k_edge_stage(const float* __restrict__ Fin, const int* __restrict__ idx,
                             const __bf16* __restrict__ wA, const float* __restrict__ sA,
                             const float* __restrict__ tA,
                             const __bf16* __restrict__ wB, const float* __restrict__ sB,
                             const float* __restrict__ tB, float* __restrict__ Xout) {
  constexpr int NT = 20;          // 320/16 column tiles
  constexpr int NKA = KA / 32;    // k-chunks, layer 1
  extern __shared__ __align__(32) char smem[];
  __bf16* featS = (__bf16*)smem;                    // [NKA*NT][32][16] frag order
  __bf16* mid   = featS + (size_t)KA * 320;         // [2*NT][32][16]  frag order
  float*  out2  = (float*)(mid + (size_t)64 * 320); // [64][320] plain

  const int b = blockIdx.y, n0 = blockIdx.x * 16;
  const int tid = threadIdx.x;
  const int lane = tid & 31, wave = tid >> 5, nwave = blockDim.x >> 5;
  const float* Fb = Fin + (size_t)b * CIN * NPTS;

  for (int i = tid; i < KA * 320; i += blockDim.x) {
    int c = i / 320, j = i % 320;         // c: channel row, j: nk column
    int p = j / KNN, kk = j % KNN;
    int n = n0 + p;
    float v = 0.f;
    if (c < CIN) {
      int nb = idx[((size_t)b * NPTS + n) * KNN + kk];
      v = Fb[c * NPTS + nb] - Fb[c * NPTS + n];
    } else if (c < 2 * CIN) {
      v = Fb[(c - CIN) * NPTS + n];
    }
    featS[b_pos((c >> 5) * NT + (j >> 4), j & 15, c & 31)] = (__bf16)v;
  }
  __syncthreads();

  const int h = lane >> 4, nl = lane & 15;

  for (int job = wave; job < 4 * NT; job += nwave) {  // 64/16 x 320/16 tiles
    int mt = job / NT, nt = job % NT;
    v8f acc = {};
#pragma unroll
    for (int kc = 0; kc < NKA; ++kc) {
      v16bf a  = load_frag16(wA, mt * NKA + kc);
      v16bf bb = load_frag16(featS, kc * NT + nt);
      acc = __builtin_amdgcn_wmma_f32_16x16x32_bf16(false, a, false, bb, (short)0, acc,
                                                    false, false);
    }
    int col = nt * 16 + nl;
#pragma unroll
    for (int r = 0; r < 8; ++r) {
      int m = mt * 16 + r + 8 * h;
      float v = lrelu(acc[r] * sA[m] + tA[m]);
      if (TWO) mid[b_pos((m >> 5) * NT + nt, nl, m & 31)] = (__bf16)v;
      else     out2[m * 320 + col] = v;
    }
  }
  __syncthreads();

  if (TWO) {
    for (int job = wave; job < 4 * NT; job += nwave) {
      int mt = job / NT, nt = job % NT;
      v8f acc = {};
#pragma unroll
      for (int kc = 0; kc < 2; ++kc) {
        v16bf a  = load_frag16(wB, mt * 2 + kc);
        v16bf bb = load_frag16(mid, kc * NT + nt);
        acc = __builtin_amdgcn_wmma_f32_16x16x32_bf16(false, a, false, bb, (short)0, acc,
                                                      false, false);
      }
      int col = nt * 16 + nl;
#pragma unroll
      for (int r = 0; r < 8; ++r) {
        int m = mt * 16 + r + 8 * h;
        out2[m * 320 + col] = lrelu(acc[r] * sB[m] + tB[m]);
      }
    }
    __syncthreads();
  }

  for (int i = tid; i < 64 * 16; i += blockDim.x) {
    int ch = i / 16, p = i % 16;
    float mx = -3e38f;
    for (int kk = 0; kk < KNN; ++kk) mx = fmaxf(mx, out2[ch * 320 + p * KNN + kk]);
    Xout[((size_t)b * 64 + ch) * NPTS + n0 + p] = mx;
  }
}

// ---------------- e = lrelu(s6*(w6 . cat)+t6); g = max over n (atomicMax) ----------
__global__ void k_w6_max(const float* __restrict__ x1, const float* __restrict__ x2,
                         const float* __restrict__ x3,
                         const __bf16* __restrict__ w6, const float* __restrict__ s6,
                         const float* __restrict__ t6, unsigned* __restrict__ g_enc) {
  extern __shared__ __align__(32) char smem[];
  __bf16* col = (__bf16*)smem;  // [6][32][16] frag order (192 rows, one 16-col tile)
  const int b = blockIdx.y, n0 = blockIdx.x * 16;
  const int tid = threadIdx.x;
  const int lane = tid & 31, wave = tid >> 5, nwave = blockDim.x >> 5;

  for (int i = tid; i < 192 * 16; i += blockDim.x) {
    int r = i >> 4, p = i & 15;
    const float* src = (r < 64) ? x1 : (r < 128) ? x2 : x3;
    int c = (r < 64) ? r : (r < 128) ? r - 64 : r - 128;
    float v = src[((size_t)b * 64 + c) * NPTS + n0 + p];
    col[b_pos(r >> 5, p, r & 31)] = (__bf16)v;
  }
  __syncthreads();

  const int h = lane >> 4, nl = lane & 15;
  for (int mt = wave; mt < 64; mt += nwave) {  // 1024/16 M-tiles
    v8f acc = {};
#pragma unroll
    for (int kc = 0; kc < 6; ++kc) {           // 192/32
      v16bf a  = load_frag16(w6, mt * 6 + kc);
      v16bf bb = load_frag16(col, kc);
      acc = __builtin_amdgcn_wmma_f32_16x16x32_bf16(false, a, false, bb, (short)0, acc,
                                                    false, false);
    }
#pragma unroll
    for (int r = 0; r < 8; ++r) {
      int m = mt * 16 + r + 8 * h;
      float v = lrelu(acc[r] * s6[m] + t6[m]);
#pragma unroll
      for (int off = 8; off; off >>= 1) v = fmaxf(v, __shfl_xor(v, off, 32));
      if (nl == 0) atomicMax(&g_enc[(size_t)b * 1024 + m], fenc(v));
    }
  }
}

// ------- head: [g,x1,x2,x3](1216) -> w7(512) -> w8(256) -> w9(1) -> sigmoid -------
__global__ void k_head(const float* __restrict__ x1, const float* __restrict__ x2,
                       const float* __restrict__ x3, const unsigned* __restrict__ g_enc,
                       const __bf16* __restrict__ w7, const float* __restrict__ s7,
                       const float* __restrict__ t7,
                       const __bf16* __restrict__ w8, const float* __restrict__ s8,
                       const float* __restrict__ t8,
                       const float* __restrict__ w9, float* __restrict__ out) {
  extern __shared__ __align__(32) char smem[];
  __bf16* col = (__bf16*)smem;            // [38][32][16] frag order (1216 rows)
  __bf16* h7  = col + 1216 * 16;          // [16][32][16] frag order (512 rows)
  float*  h8  = (float*)(h7 + 512 * 16);  // [256][16] plain
  const int b = blockIdx.y, n0 = blockIdx.x * 16;
  const int tid = threadIdx.x;
  const int lane = tid & 31, wave = tid >> 5, nwave = blockDim.x >> 5;

  for (int i = tid; i < 1216 * 16; i += blockDim.x) {
    int r = i >> 4, p = i & 15;
    float v;
    if (r < 1024) v = fdec(g_enc[(size_t)b * 1024 + r]);
    else {
      const float* src = (r < 1088) ? x1 : (r < 1152) ? x2 : x3;
      int c = (r < 1088) ? r - 1024 : (r < 1152) ? r - 1088 : r - 1152;
      v = src[((size_t)b * 64 + c) * NPTS + n0 + p];
    }
    col[b_pos(r >> 5, p, r & 31)] = (__bf16)v;
  }
  __syncthreads();

  const int h = lane >> 4, nl = lane & 15;
  for (int mt = wave; mt < 32; mt += nwave) {  // 512/16
    v8f acc = {};
#pragma unroll
    for (int kc = 0; kc < 38; ++kc) {          // 1216/32
      if (kc + 1 < 38)                         // hide L2 latency on deep-K GEMM
        __builtin_prefetch(w7 + ((size_t)(mt * 38 + kc + 1) * 32) * 16, 0, 3);
      v16bf a  = load_frag16(w7, mt * 38 + kc);
      v16bf bb = load_frag16(col, kc);
      acc = __builtin_amdgcn_wmma_f32_16x16x32_bf16(false, a, false, bb, (short)0, acc,
                                                    false, false);
    }
#pragma unroll
    for (int r = 0; r < 8; ++r) {
      int m = mt * 16 + r + 8 * h;
      float v = lrelu(acc[r] * s7[m] + t7[m]);
      h7[b_pos(m >> 5, nl, m & 31)] = (__bf16)v;
    }
  }
  __syncthreads();

  for (int mt = wave; mt < 16; mt += nwave) {  // 256/16
    v8f acc = {};
#pragma unroll
    for (int kc = 0; kc < 16; ++kc) {          // 512/32
      v16bf a  = load_frag16(w8, mt * 16 + kc);
      v16bf bb = load_frag16(h7, kc);
      acc = __builtin_amdgcn_wmma_f32_16x16x32_bf16(false, a, false, bb, (short)0, acc,
                                                    false, false);
    }
#pragma unroll
    for (int r = 0; r < 8; ++r) {
      int m = mt * 16 + r + 8 * h;
      h8[m * 16 + nl] = lrelu(acc[r] * s8[m] + t8[m]);
    }
  }
  __syncthreads();

  // w9 (1x256) dot + sigmoid: 16 half-wave groups, one per point (block=256)
  {
    int p = tid >> 4;
    int t = tid & 15;
    float s = 0.f;
    for (int c = t; c < 256; c += 16) s += w9[c] * h8[c * 16 + p];
#pragma unroll
    for (int off = 8; off; off >>= 1) s += __shfl_xor(s, off, 32);
    if (t == 0) out[(size_t)b * NPTS + n0 + p] = 1.f / (1.f + expf(-s));
  }
}

// ---------------- launch ----------------
extern "C" void kernel_launch(void* const* d_in, const int* in_sizes, int n_in,
                              void* d_out, int out_size, void* d_ws, size_t ws_size,
                              hipStream_t stream) {
  (void)in_sizes; (void)n_in; (void)out_size; (void)ws_size;
  const float* x = (const float*)d_in[0];
  const float* w[10];
  for (int i = 1; i <= 9; ++i) w[i] = (const float*)d_in[i];
  const float* s_[9]; const float* t_[9];
  for (int i = 1; i <= 8; ++i) {
    s_[i] = (const float*)d_in[10 + 2 * (i - 1)];
    t_[i] = (const float*)d_in[11 + 2 * (i - 1)];
  }
  float* out = (float*)d_out;

  char* ws = (char*)d_ws;
  size_t off = 0;
  auto alloc = [&](size_t bytes) -> char* {
    char* p = ws + off;
    off += (bytes + 255) & ~(size_t)255;
    return p;
  };
  __bf16* wb1 = (__bf16*)alloc(64 * 32 * 2);  // w1 K padded 4->32
  __bf16* wb2 = (__bf16*)alloc(64 * 64 * 2);
  __bf16* wb3 = (__bf16*)alloc(64 * 128 * 2);
  __bf16* wb4 = (__bf16*)alloc(64 * 64 * 2);
  __bf16* wb5 = (__bf16*)alloc(64 * 128 * 2);
  __bf16* wb6 = (__bf16*)alloc((size_t)1024 * 192 * 2);
  __bf16* wb7 = (__bf16*)alloc((size_t)512 * 1216 * 2);
  __bf16* wb8 = (__bf16*)alloc((size_t)256 * 512 * 2);
  float* x1 = (float*)alloc((size_t)BATCH * 64 * NPTS * 4);
  float* x2 = (float*)alloc((size_t)BATCH * 64 * NPTS * 4);
  float* x3 = (float*)alloc((size_t)BATCH * 64 * NPTS * 4);
  int*   idx = (int*)alloc((size_t)BATCH * NPTS * KNN * 4);
  unsigned* g_enc = (unsigned*)alloc((size_t)BATCH * 1024 * 4);
  float* xswz = (float*)alloc((size_t)BATCH * (NPTS / 16) * 16 * 64 * 4);  // max NKC=16
  float* xx_g = (float*)alloc((size_t)BATCH * NPTS * 4);

  dim3 blk(256);
  auto cvt = [&](const float* src, __bf16* dst, int O, int C, int Cpad) {
    int n = O * Cpad;
    k_convert_w_swz<<<(n + 255) / 256, blk, 0, stream>>>(src, dst, O, C, Cpad);
  };
  cvt(w[1], wb1, 64, 4, 32);
  cvt(w[2], wb2, 64, 64, 64);
  cvt(w[3], wb3, 64, 128, 128);
  cvt(w[4], wb4, 64, 64, 64);
  cvt(w[5], wb5, 64, 128, 128);
  cvt(w[6], wb6, 1024, 192, 192);
  cvt(w[7], wb7, 512, 1216, 1216);
  cvt(w[8], wb8, 256, 512, 512);
  k_init_g<<<(BATCH * 1024 + 255) / 256, blk, 0, stream>>>(g_enc);

  dim3 grid(NPTS / 16, BATCH);
  const size_t knnLds = (size_t)(16 * NPTS + 16) * sizeof(float);
  auto stageLds = [](int KA) -> size_t {
    return (size_t)KA * 320 * 2 + (size_t)64 * 320 * 2 + (size_t)64 * 320 * 4;
  };
  const int xxBlocks = (BATCH * NPTS + 255) / 256;

  // stage 1: C=2 (NKC=1), convs w1->w2
  {
    constexpr int PER = (NPTS / 16) * 1 * 64;
    k_swz_x<2><<<dim3((PER + 255) / 256, BATCH), blk, 0, stream>>>(x, xswz);
    k_xx<<<xxBlocks, blk, 0, stream>>>(x, 2, xx_g);
    k_knn<2><<<grid, blk, knnLds, stream>>>(xswz, xx_g, idx);
    k_edge_stage<2, 32, true><<<grid, blk, stageLds(32), stream>>>(
        x, idx, wb1, s_[1], t_[1], wb2, s_[2], t_[2], x1);
  }
  // stage 2: C=64, convs w3->w4
  {
    constexpr int PER = (NPTS / 16) * 16 * 64;
    k_swz_x<64><<<dim3((PER + 255) / 256, BATCH), blk, 0, stream>>>(x1, xswz);
    k_xx<<<xxBlocks, blk, 0, stream>>>(x1, 64, xx_g);
    k_knn<64><<<grid, blk, knnLds, stream>>>(xswz, xx_g, idx);
    k_edge_stage<64, 128, true><<<grid, blk, stageLds(128), stream>>>(
        x1, idx, wb3, s_[3], t_[3], wb4, s_[4], t_[4], x2);
  }
  // stage 3: C=64, single conv w5
  {
    constexpr int PER = (NPTS / 16) * 16 * 64;
    k_swz_x<64><<<dim3((PER + 255) / 256, BATCH), blk, 0, stream>>>(x2, xswz);
    k_xx<<<xxBlocks, blk, 0, stream>>>(x2, 64, xx_g);
    k_knn<64><<<grid, blk, knnLds, stream>>>(xswz, xx_g, idx);
    k_edge_stage<64, 128, false><<<grid, blk, stageLds(128), stream>>>(
        x2, idx, wb5, s_[5], t_[5], nullptr, nullptr, nullptr, x3);
  }

  k_w6_max<<<grid, blk, (size_t)192 * 16 * 2, stream>>>(x1, x2, x3, wb6, s_[6], t_[6], g_enc);
  k_head<<<grid, blk, (size_t)1216 * 16 * 2 + 512 * 16 * 2 + 256 * 16 * 4, stream>>>(
      x1, x2, x3, g_enc, wb7, s_[7], t_[7], wb8, s_[8], t_[8], w[9], out);
}
